// BackboneCurvatureMixer_55018531062584
// MI455X (gfx1250) — compile-verified, matrix operand order checked by reading, not run on hardware
//
#include <hip/hip_runtime.h>
#include <cstdint>

// ---------------------------------------------------------------------------
// Problem constants
// ---------------------------------------------------------------------------
constexpr int B_   = 8;
constexpr int L_   = 4096;
constexpr int D_   = 1024;
constexpr int R_   = 32;
constexpr int HID_ = 512;
constexpr int PLU_ = 496;            // 32*31/2
constexpr int M_   = B_ * L_;        // 32768 rows
constexpr int P3_  = 3 * PLU_;       // 1488
constexpr int P3P_ = 1504;           // padded to mult of 32
constexpr int PLUP_= 512;            // padded to mult of 32

// Output layout (flat, return order: z, g_bb, g_curv, p_bb1, kappa)
constexpr size_t OFF_Z   = 0;
constexpr size_t OFF_GBB = (size_t)M_ * R_;                 // 1,048,576
constexpr size_t OFF_GCV = OFF_GBB + (size_t)M_ * D_;       // 34,603,008
constexpr size_t OFF_PBB = OFF_GCV + (size_t)M_ * D_;       // 68,157,440
constexpr size_t OFF_KAP = OFF_PBB + (size_t)M_ * PLU_;     // 84,410,368

// Workspace layout (bytes, 256-aligned)
constexpr size_t WS_PALL   = 0;                                   // M*1504 f16
constexpr size_t WS_KAPPA  = WS_PALL  + (size_t)M_ * P3P_ * 2;    // M*512 f16
constexpr size_t WS_ACTBB  = WS_KAPPA + (size_t)M_ * PLUP_ * 2;   // M*512 f16
constexpr size_t WS_W1BB   = WS_ACTBB + (size_t)M_ * HID_ * 2;    // 512*1504 f16
constexpr size_t WS_W1CV   = WS_W1BB  + (size_t)HID_ * P3P_ * 2;  // 512*512 f16
constexpr size_t WS_W2BB   = WS_W1CV  + (size_t)HID_ * PLUP_ * 2; // 1024*512 f16
constexpr size_t WS_W2CV   = WS_W2BB  + (size_t)D_ * HID_ * 2;    // 1024*512 f16
// act_cv aliases WS_PALL (p_all dead after bb layer-1 GEMM)
constexpr size_t WS_ACTCV  = WS_PALL;

// ---------------------------------------------------------------------------
// CDNA5 async global->LDS (ASYNCcnt-tracked) with graceful fallback
// ---------------------------------------------------------------------------
#if defined(__has_builtin)
#  if __has_builtin(__builtin_amdgcn_global_load_async_to_lds_b128)
#    define HAVE_ASYNC_LDS 1
#  else
#    define HAVE_ASYNC_LDS 0
#  endif
#  if __has_builtin(__builtin_amdgcn_s_wait_asynccnt)
#    define HAVE_WAIT_ASYNC 1
#  else
#    define HAVE_WAIT_ASYNC 0
#  endif
#else
#  define HAVE_ASYNC_LDS 0
#  define HAVE_WAIT_ASYNC 0
#endif

typedef __attribute__((ext_vector_type(4))) int v4i;
typedef __attribute__((address_space(1))) v4i* gas_v4i;   // global
typedef __attribute__((address_space(3))) v4i* las_v4i;   // LDS

__device__ __forceinline__ void async_copy16(const _Float16* g, _Float16* l) {
#if HAVE_ASYNC_LDS
    __builtin_amdgcn_global_load_async_to_lds_b128(
        (gas_v4i)(_Float16*)g, (las_v4i)l, 0, 0);
#else
    *(uint4*)l = *(const uint4*)g;
#endif
}

__device__ __forceinline__ void wait_async0() {
#if HAVE_ASYNC_LDS
#  if HAVE_WAIT_ASYNC
    __builtin_amdgcn_s_wait_asynccnt(0);
#  else
    asm volatile("s_wait_asynccnt 0x0" ::: "memory");
#  endif
#endif
}

// ---------------------------------------------------------------------------
// WMMA helpers (wave32, 16x16x32 f16 -> f32)
// ---------------------------------------------------------------------------
typedef __attribute__((ext_vector_type(16))) _Float16 v16h;
typedef __attribute__((ext_vector_type(8)))  float    v8f;

union Frag16 { v16h v; unsigned u[8]; };

__device__ __forceinline__ unsigned pack2h(float a, float b) {
    union { _Float16 h[2]; unsigned u; } t;
    t.h[0] = (_Float16)a; t.h[1] = (_Float16)b;
    return t.u;
}

__device__ __forceinline__ v8f wmma_f16(const Frag16& a, const Frag16& b, v8f c) {
    return __builtin_amdgcn_wmma_f32_16x16x32_f16(
        /*neg_a=*/false, a.v, /*neg_b=*/false, b.v,
        /*c_mod=*/(short)0, c, /*reuse_a=*/false, /*reuse_b=*/false);
}

// 16-bit A/B operand K-offset pattern (ISA 7.12.2): per lane the fragment is
// two contiguous 8-half chunks: K in [hi, hi+8) (v0..v3) and [16+hi, 16+hi+8)
// (v4..v7), where hi = (lane>=16)*8.
__device__ __forceinline__ int kpat(int j, int hi) {
    return (j < 4) ? (hi + 2 * j) : (16 + hi + 2 * (j - 4));
}

__device__ __forceinline__ float gelu_exact(float x) {
    return 0.5f * x * (1.0f + erff(x * 0.70710678118654752f));
}

// ---------------------------------------------------------------------------
// Kernel: f32 -> f16 weight conversion with K padding (zeros in pad columns)
// ---------------------------------------------------------------------------
__global__ void k_w2half(const float* __restrict__ src, _Float16* __restrict__ dst,
                         int N, int K, int Kpad) {
    size_t total = (size_t)N * Kpad;
    for (size_t i = (size_t)blockIdx.x * blockDim.x + threadIdx.x; i < total;
         i += (size_t)gridDim.x * blockDim.x) {
        int n = (int)(i / Kpad);
        int k = (int)(i % Kpad);
        float v = (k < K) ? src[(size_t)n * K + k] : 0.0f;
        dst[i] = (_Float16)v;
    }
}

// ---------------------------------------------------------------------------
// Kernel: z = h @ w_red^T + b_red   (M x 32, K = 1024)
// 8 waves/block, each wave: 16 rows x 32 cols (two 16x16 accumulators)
// h and w_red converted f32->f16 in registers; f32 accumulate via WMMA.
// ---------------------------------------------------------------------------
__global__ void __launch_bounds__(256)
k_z(const float* __restrict__ h, const float* __restrict__ w_red,
    const float* __restrict__ b_red, float* __restrict__ z) {
    const int tid = threadIdx.x, lane = tid & 31, wave = tid >> 5;
    const int rowBase = blockIdx.x * 128 + wave * 16;
    const int hi = (lane >> 4) * 8, mloc = lane & 15;

    v8f c0 = {}; v8f c1 = {};
    const size_t hrow = (size_t)(rowBase + mloc) * D_;
    const size_t w0   = (size_t)mloc * D_;
    const size_t w1   = (size_t)(mloc + 16) * D_;

    for (int k0 = 0; k0 < D_; k0 += 32) {
        Frag16 a, b0, b1;
#pragma unroll
        for (int j = 0; j < 8; ++j) {
            const int ko = k0 + kpat(j, hi);
            float2 fa = *(const float2*)&h[hrow + ko];
            a.u[j]  = pack2h(fa.x, fa.y);
            float2 fb = *(const float2*)&w_red[w0 + ko];
            b0.u[j] = pack2h(fb.x, fb.y);
            float2 fc = *(const float2*)&w_red[w1 + ko];
            b1.u[j] = pack2h(fc.x, fc.y);
        }
        c0 = wmma_f16(a, b0, c0);
        c1 = wmma_f16(a, b1, c1);
    }

    const int msel = (lane >> 4) * 8;
    const float bias0 = b_red[mloc], bias1 = b_red[mloc + 16];
#pragma unroll
    for (int e = 0; e < 8; ++e) {
        const int m = rowBase + msel + e;
        z[(size_t)m * R_ + mloc]      = c0[e] + bias0;
        z[(size_t)m * R_ + 16 + mloc] = c1[e] + bias1;
    }
}

// ---------------------------------------------------------------------------
// Kernel: Plücker features. One block per row (b,l).
//   p_d[k] = u[ia]*v[ib] - u[ib]*v[ia], normalized per (row,delta), masked.
// Writes p_all (f16, stride 1504, pad zeroed) and p_bb1 (f32 output).
// ---------------------------------------------------------------------------
__global__ void __launch_bounds__(256)
k_plucker(const float* __restrict__ z, const unsigned char* __restrict__ mask,
          _Float16* __restrict__ p_all, float* __restrict__ p_bb1) {
    __shared__ float zl[4][R_];     // u, v(+1), v(+2), v(+4)
    __shared__ float pbuf[PLU_];
    __shared__ float ssum;

    const int row = blockIdx.x;
    const int l   = row & (L_ - 1);
    const int tid = threadIdx.x;
    const int offs[4] = {0, 1, 2, 4};

    if (tid < 128) {
        const int d = tid >> 5, r = tid & 31;
        const int ll = l + offs[d];
        zl[d][r] = (ll < L_) ? z[(size_t)(row + offs[d]) * R_ + r] : 0.0f;
    }
    const float mu = (mask[row] != 0) ? 1.0f : 0.0f;
    __syncthreads();

    for (int dI = 0; dI < 3; ++dI) {
        const int delta = offs[dI + 1];
        if (tid == 0) ssum = 0.0f;
        __syncthreads();

        float part = 0.0f;
        for (int k = tid; k < PLU_; k += 256) {
            int ia = 0, rem = k, span = R_ - 1;
            while (rem >= span) { rem -= span; ++ia; --span; }
            const int ib = ia + 1 + rem;
            const float p = zl[0][ia] * zl[1 + dI][ib] - zl[0][ib] * zl[1 + dI][ia];
            pbuf[k] = p;
            part += p * p;
        }
        atomicAdd(&ssum, part);          // ds_add_f32 on LDS
        __syncthreads();

        const float mv = (l + delta < L_) ? ((mask[row + delta] != 0) ? 1.0f : 0.0f) : 0.0f;
        const float scale = mu * mv / fmaxf(sqrtf(ssum), 1e-8f);

        for (int k = tid; k < PLU_; k += 256) {
            const float v = pbuf[k] * scale;
            p_all[(size_t)row * P3P_ + dI * PLU_ + k] = (_Float16)v;
            if (dI == 0) p_bb1[(size_t)row * PLU_ + k] = v;
        }
        __syncthreads();
    }
    if (tid < P3P_ - P3_) p_all[(size_t)row * P3P_ + P3_ + tid] = (_Float16)0.0f;
}

// ---------------------------------------------------------------------------
// Kernel: kappa = (p[l+1] - 2 p[l] + p[l-1]) * mask  (f32 out + f16 ws, padded)
// ---------------------------------------------------------------------------
__global__ void k_kappa(const float* __restrict__ p_bb1,
                        const unsigned char* __restrict__ mask,
                        float* __restrict__ kappa, _Float16* __restrict__ kappa16) {
    const size_t total = (size_t)M_ * PLU_;
    for (size_t idx = (size_t)blockIdx.x * blockDim.x + threadIdx.x; idx < total;
         idx += (size_t)gridDim.x * blockDim.x) {
        const size_t row = idx / PLU_;
        const int    k   = (int)(idx % PLU_);
        const int    l   = (int)(row & (L_ - 1));
        const float  c   = p_bb1[row * PLU_ + k];
        const float  f   = (l + 1 < L_) ? p_bb1[(row + 1) * PLU_ + k] : 0.0f;
        const float  b   = (l > 0)      ? p_bb1[(row - 1) * PLU_ + k] : 0.0f;
        const float  m   = (mask[row] != 0) ? 1.0f : 0.0f;
        const float  v   = (f - 2.0f * c + b) * m;
        kappa[row * PLU_ + k]    = v;
        kappa16[row * PLUP_ + k] = (_Float16)v;
    }
    // zero the 16 pad columns
    const size_t padTotal = (size_t)M_ * (PLUP_ - PLU_);
    for (size_t idx = (size_t)blockIdx.x * blockDim.x + threadIdx.x; idx < padTotal;
         idx += (size_t)gridDim.x * blockDim.x) {
        const size_t row = idx >> 4;
        kappa16[row * PLUP_ + PLU_ + (idx & 15)] = (_Float16)0.0f;
    }
}

// ---------------------------------------------------------------------------
// Kernel: tiled WMMA GEMM  C[M x N] = act(A[M x K] @ W[N x K]^T + bias)
// Block tile 128x64, 8 waves as 4x2, wave tile 32x32 (2x2 WMMA frags).
// Double-buffered LDS; tiles staged with GLOBAL_LOAD_ASYNC_TO_LDS_B128 when
// available (ASYNCcnt), so WMMA on tile k overlaps the fetch of tile k+1.
// ---------------------------------------------------------------------------
template <bool GELU, bool OUT16>
__global__ void __launch_bounds__(256)
k_gemm(const _Float16* __restrict__ A, int lda,
       const _Float16* __restrict__ W, int ldw,
       const float* __restrict__ bias,
       float* __restrict__ outF, _Float16* __restrict__ out16, int ldo,
       int Ktiles) {
    __shared__ _Float16 sA[2][128][32];
    __shared__ _Float16 sB[2][64][32];

    const int tid  = threadIdx.x;
    const int lane = tid & 31, wave = tid >> 5;
    const int wm = wave >> 1;         // 0..3
    const int wn = wave & 1;          // 0..1
    const int mBlock = blockIdx.x * 128;
    const int nBlock = blockIdx.y * 64;

    v8f acc[2][2] = {{{}, {}}, {{}, {}}};

    const int hi   = (lane >> 4) * 8;
    const int mloc = lane & 15;

    // Tile staging: A = 512 x 16B (2 per thread), B = 256 x 16B (1 per thread)
    const int arow0 = (tid) >> 2,          ac0 = ((tid) & 3) * 8;
    const int arow1 = (tid + 256) >> 2,    ac1 = ((tid + 256) & 3) * 8;
    const int brow  = tid >> 2,            bc0 = (tid & 3) * 8;

    auto stage = [&](int kt, int buf) {
        const int k0 = kt * 32;
        async_copy16(A + (size_t)(mBlock + arow0) * lda + k0 + ac0, &sA[buf][arow0][ac0]);
        async_copy16(A + (size_t)(mBlock + arow1) * lda + k0 + ac1, &sA[buf][arow1][ac1]);
        async_copy16(W + (size_t)(nBlock + brow) * ldw + k0 + bc0,  &sB[buf][brow][bc0]);
    };

    stage(0, 0);
    wait_async0();
    __syncthreads();

    for (int kt = 0; kt < Ktiles; ++kt) {
        const int cur = kt & 1;
        if (kt + 1 < Ktiles) stage(kt + 1, cur ^ 1);
        if (kt + 2 < Ktiles) {
            // far prefetch of the streamed operand (global_prefetch_b8)
            __builtin_prefetch(A + (size_t)(mBlock + arow0) * lda + (kt + 2) * 32, 0, 1);
        }

        // Operand fragments: two contiguous ds_load_b128 per matrix per tile
        Frag16 af[2], bf[2];
#pragma unroll
        for (int t = 0; t < 2; ++t) {
            const int mrow = wm * 32 + t * 16 + mloc;
            const int nrow = wn * 32 + t * 16 + mloc;
            *(uint4*)&af[t].u[0] = *(const uint4*)&sA[cur][mrow][hi];
            *(uint4*)&af[t].u[4] = *(const uint4*)&sA[cur][mrow][16 + hi];
            *(uint4*)&bf[t].u[0] = *(const uint4*)&sB[cur][nrow][hi];
            *(uint4*)&bf[t].u[4] = *(const uint4*)&sB[cur][nrow][16 + hi];
        }
#pragma unroll
        for (int i = 0; i < 2; ++i)
#pragma unroll
            for (int j = 0; j < 2; ++j)
                acc[i][j] = wmma_f16(af[i], bf[j], acc[i][j]);

        wait_async0();     // next tile's async writes complete (after compute)
        __syncthreads();
    }

    // Epilogue: C frag element e -> (M = e + 8*(lane>=16), N = lane&15)
    const int nloc = lane & 15;
    const int msel = (lane >> 4) * 8;
#pragma unroll
    for (int i = 0; i < 2; ++i) {
        const int mBase = mBlock + wm * 32 + i * 16 + msel;
#pragma unroll
        for (int j = 0; j < 2; ++j) {
            const int n = nBlock + wn * 32 + j * 16 + nloc;
            const float bia = bias[n];
#pragma unroll
            for (int e = 0; e < 8; ++e) {
                float v = acc[i][j][e] + bia;
                if (GELU) v = gelu_exact(v);
                const size_t o = (size_t)(mBase + e) * ldo + n;
                if (OUT16) out16[o] = (_Float16)v;
                else       outF[o]  = v;
            }
        }
    }
}

// ---------------------------------------------------------------------------
// Host launcher
// ---------------------------------------------------------------------------
extern "C" void kernel_launch(void* const* d_in, const int* in_sizes, int n_in,
                              void* d_out, int out_size, void* d_ws, size_t ws_size,
                              hipStream_t stream) {
    const float*         h        = (const float*)d_in[0];
    const unsigned char* seq_mask = (const unsigned char*)d_in[1];
    const float*         w_red    = (const float*)d_in[2];
    const float*         b_red    = (const float*)d_in[3];
    const float*         bb_w1    = (const float*)d_in[4];
    const float*         bb_b1    = (const float*)d_in[5];
    const float*         bb_w2    = (const float*)d_in[6];
    const float*         bb_b2    = (const float*)d_in[7];
    const float*         cv_w1    = (const float*)d_in[8];
    const float*         cv_b1    = (const float*)d_in[9];
    const float*         cv_w2    = (const float*)d_in[10];
    const float*         cv_b2    = (const float*)d_in[11];

    float* out     = (float*)d_out;
    float* z_out   = out + OFF_Z;
    float* gbb_out = out + OFF_GBB;
    float* gcv_out = out + OFF_GCV;
    float* pbb_out = out + OFF_PBB;
    float* kap_out = out + OFF_KAP;

    char* ws = (char*)d_ws;
    _Float16* p_all   = (_Float16*)(ws + WS_PALL);
    _Float16* kappa16 = (_Float16*)(ws + WS_KAPPA);
    _Float16* act_bb  = (_Float16*)(ws + WS_ACTBB);
    _Float16* act_cv  = (_Float16*)(ws + WS_ACTCV);   // aliases p_all (dead by then)
    _Float16* w1bb    = (_Float16*)(ws + WS_W1BB);
    _Float16* w1cv    = (_Float16*)(ws + WS_W1CV);
    _Float16* w2bb    = (_Float16*)(ws + WS_W2BB);
    _Float16* w2cv    = (_Float16*)(ws + WS_W2CV);

    // 1) weights -> f16 (K padded to mult of 32)
    k_w2half<<<1024, 256, 0, stream>>>(bb_w1, w1bb, HID_, P3_, P3P_);
    k_w2half<<<1024, 256, 0, stream>>>(cv_w1, w1cv, HID_, PLU_, PLUP_);
    k_w2half<<<1024, 256, 0, stream>>>(bb_w2, w2bb, D_, HID_, HID_);
    k_w2half<<<1024, 256, 0, stream>>>(cv_w2, w2cv, D_, HID_, HID_);

    // 2) z = h @ w_red^T + b_red
    k_z<<<M_ / 128, 256, 0, stream>>>(h, w_red, b_red, z_out);

    // 3) Plücker features -> p_bb1 (out) + p_all (ws, f16)
    k_plucker<<<M_, 256, 0, stream>>>(z_out, seq_mask, p_all, pbb_out);

    // 4) kappa -> out (f32) + ws (f16)
    k_kappa<<<4096, 256, 0, stream>>>(pbb_out, seq_mask, kap_out, kappa16);

    // 5) backbone MLP
    {
        dim3 g1(M_ / 128, HID_ / 64);
        k_gemm<true, true><<<g1, 256, 0, stream>>>(
            p_all, P3P_, w1bb, P3P_, bb_b1, nullptr, act_bb, HID_, P3P_ / 32);
        dim3 g2(M_ / 128, D_ / 64);
        k_gemm<false, false><<<g2, 256, 0, stream>>>(
            act_bb, HID_, w2bb, HID_, bb_b2, gbb_out, nullptr, D_, HID_ / 32);
    }

    // 6) curvature MLP
    {
        dim3 g1(M_ / 128, HID_ / 64);
        k_gemm<true, true><<<g1, 256, 0, stream>>>(
            kappa16, PLUP_, w1cv, PLUP_, cv_b1, nullptr, act_cv, HID_, PLUP_ / 32);
        dim3 g2(M_ / 128, D_ / 64);
        k_gemm<false, false><<<g2, 256, 0, stream>>>(
            act_cv, HID_, w2cv, HID_, cv_b2, gcv_out, nullptr, D_, HID_ / 32);
    }
}
